// VisionEncoder_18640158065315
// MI455X (gfx1250) — matrix operand order repaired and checked by reading, not compiled, not run on hardware
//
#include <hip/hip_runtime.h>

// ---------------------------------------------------------------------------
// Types for CDNA5 WMMA (wave32): bf16 A/B fragments, f32 accumulator
// ---------------------------------------------------------------------------
typedef __attribute__((ext_vector_type(16))) __bf16 v16bf;
typedef __attribute__((ext_vector_type(8)))  __bf16 v8bf;
typedef __attribute__((ext_vector_type(8)))  float  v8f;
typedef int v4i __attribute__((ext_vector_type(4)));

#define BN_RSQRT 0.9999950000374998f   // 1/sqrt(1 + 1e-5)

// gfx1250 async global->LDS path (guarded: degrades to sync copy if absent)
#if defined(__gfx1250__) && __has_builtin(__builtin_amdgcn_global_load_async_to_lds_b128)
#define HAVE_ASYNC 1
#define AS_GLOBAL __attribute__((address_space(1)))
#define AS_LDS    __attribute__((address_space(3)))
#else
#define HAVE_ASYNC 0
#endif

__device__ __forceinline__ void wait_async_then_barrier()
{
#if HAVE_ASYNC
#if __has_builtin(__builtin_amdgcn_s_wait_asynccnt)
    __builtin_amdgcn_s_wait_asynccnt(0);
#else
    asm volatile("s_wait_asynccnt 0x0" ::: "memory");
#endif
#endif
    __syncthreads();
}

// ---------------------------------------------------------------------------
// Weight pre-convert: fp32 [Cout,K] -> bf16 [Cout,Kp] zero-padded (Kp = K up to /32)
// ---------------------------------------------------------------------------
__global__ __launch_bounds__(256)
void convert_weights(const float* __restrict__ w, __bf16* __restrict__ wb,
                     int Cout, int K, int Kp)
{
    long i = (long)blockIdx.x * 256 + threadIdx.x;
    long total = (long)Cout * Kp;
    if (i >= total) return;
    int co = (int)(i / Kp), k = (int)(i % Kp);
    wb[i] = (k < K) ? (__bf16)w[(size_t)co * K + k] : (__bf16)0.f;
}

// ---------------------------------------------------------------------------
// conv1: fused bilinear resize + canvas tiling + 5x5/s3/p2 conv (3->16) + BN + ReLU
// imgs: [6,B,3,450,800] f32 ; out: [B,16,200,600] bf16
// ---------------------------------------------------------------------------
__device__ __forceinline__ float sample_canvas(const float* __restrict__ imgs,
                                               int B, int b, int ci, int ih, int iw)
{
    int rb = (ih >= 300) ? 1 : 0;
    int y  = ih - rb * 300;
    int cb = (iw >= 1200) ? 2 : ((iw >= 600) ? 1 : 0);
    int x  = iw - cb * 600;
    int cam = rb * 3 + (cb == 0 ? 1 : (cb == 1 ? 0 : 2));   // {{1,0,2},{4,3,5}}
    float fy = (y + 0.5f) * 1.5f - 0.5f;
    float fx = (x + 0.5f) * (4.0f / 3.0f) - 0.5f;
    int y0 = (int)fy, x0 = (int)fx;
    float ty = fy - (float)y0, tx = fx - (float)x0;
    int y1 = (y0 + 1 < 450) ? y0 + 1 : 449;
    int x1 = (x0 + 1 < 800) ? x0 + 1 : 799;
    const float* p = imgs + ((((size_t)cam * B + b) * 3 + ci) * 450) * 800;
    float v00 = p[y0 * 800 + x0], v01 = p[y0 * 800 + x1];
    float v10 = p[y1 * 800 + x0], v11 = p[y1 * 800 + x1];
    return (1.f - ty) * ((1.f - tx) * v00 + tx * v01)
         +        ty  * ((1.f - tx) * v10 + tx * v11);
}

__global__ __launch_bounds__(256)
void conv1_fused(const float* __restrict__ imgs,
                 const float* __restrict__ w,     // [16,3,5,5]
                 const float* __restrict__ bias,
                 const float* __restrict__ g, const float* __restrict__ be,
                 __bf16* __restrict__ out, int B)
{
    __shared__ float lWT[75 * 16];    // transposed: [k][co]
    __shared__ float lBias[16], lScale[16], lBeta[16];
    for (int i = threadIdx.x; i < 75 * 16; i += 256) {
        int k = i >> 4, co = i & 15;
        lWT[i] = w[co * 75 + k];
    }
    if (threadIdx.x < 16) {
        lBias[threadIdx.x]  = bias[threadIdx.x];
        lScale[threadIdx.x] = g[threadIdx.x] * BN_RSQRT;
        lBeta[threadIdx.x]  = be[threadIdx.x];
    }
    __syncthreads();

    long gid = (long)blockIdx.x * 256 + threadIdx.x;
    const long NPOS = (long)B * 200 * 600;
    if (gid >= NPOS) return;
    int ow = (int)(gid % 600);
    int t  = (int)(gid / 600);
    int oh = t % 200;
    int b  = t / 200;

    float acc[16];
#pragma unroll
    for (int co = 0; co < 16; ++co) acc[co] = 0.f;

    for (int ci = 0; ci < 3; ++ci) {
        for (int ky = 0; ky < 5; ++ky) {
            int ih = oh * 3 - 2 + ky;
            if (ih < 0 || ih >= 600) continue;
            for (int kx = 0; kx < 5; ++kx) {
                int iw = ow * 3 - 2 + kx;
                if (iw < 0 || iw >= 1800) continue;
                float v = sample_canvas(imgs, B, b, ci, ih, iw);
                const float* wr = &lWT[(ci * 25 + ky * 5 + kx) * 16];
#pragma unroll
                for (int co = 0; co < 16; ++co) acc[co] = fmaf(v, wr[co], acc[co]);
            }
        }
    }
#pragma unroll
    for (int co = 0; co < 16; ++co) {
        float v = (acc[co] + lBias[co]) * lScale[co] + lBeta[co];
        v = fmaxf(v, 0.f);
        out[(((size_t)b * 16 + co) * 200 + oh) * 600 + ow] = (__bf16)v;
    }
}

// ---------------------------------------------------------------------------
// Implicit-GEMM conv via V_WMMA_F32_16X16X32_BF16, compile-time shapes.
//   D[m,n] = sum_k W[m,k] * im2col[k,n]
// A tile: async global->LDS b128 copies of pre-converted bf16 weights.
// B tile: im2col gather, stored n-major so fragments are contiguous 32B runs.
// ---------------------------------------------------------------------------
template <int MTILE, int NTILE, bool HAS_BN, bool OUT_BF16,
          int CIN, int HIN, int WIN, int COUT, int HOUT, int WOUT>
__global__ __launch_bounds__((MTILE / 16) * (NTILE / 16) * 32)
void conv_wmma(const __bf16* __restrict__ in,    // [B,CIN,HIN,WIN]
               const __bf16* __restrict__ wb,    // [COUT, KP] bf16, zero-padded
               const float* __restrict__ bias,
               const float* __restrict__ bn_g, const float* __restrict__ bn_b,
               void* __restrict__ outp,          // [B,COUT,HOUT,WOUT]
               int B)
{
    constexpr int WM = MTILE / 16, WN = NTILE / 16;
    constexpr int NT = WM * WN * 32;
    constexpr int K  = CIN * 25;
    constexpr int KP = (K + 31) & ~31;
    static_assert(MTILE * 4 == NT, "A-tile async copy assumes 16B per thread");
    static_assert(COUT % MTILE == 0, "no M bounds checks");
    const int N = B * HOUT * WOUT;

    __shared__ alignas(64) __bf16 lA[MTILE * 32];   // [m][k]  (row = 64B)
    __shared__ alignas(64) __bf16 lB[NTILE * 32];   // [n][k]  transposed (row = 64B)

    const int tid   = threadIdx.x;
    const int nBase = blockIdx.x * NTILE;
    const int mBase = blockIdx.y * MTILE;
    const int wave  = tid >> 5, lane = tid & 31;
    const int wn = wave % WN, wm = wave / WN;
    const int hl  = lane >> 4;
    const int l15 = lane & 15;

    // per-thread A-copy coordinates: 16B chunk q of row m
    const int am = tid >> 2, aq = tid & 3;

    v8f acc = {};

    for (int k0 = 0; k0 < KP; k0 += 32) {
        // ---- stage A tile: bf16 weights, contiguous 16B per thread ----
        {
            const __bf16* src = wb + (size_t)(mBase + am) * KP + k0 + aq * 8;
            __bf16* dst = &lA[am * 32 + aq * 8];
#if HAVE_ASYNC
            __builtin_amdgcn_global_load_async_to_lds_b128(
                (AS_GLOBAL v4i*)src, (AS_LDS v4i*)dst, 0, 0);
#else
            *(v8bf*)dst = *(const v8bf*)src;
#endif
        }
        // ---- stage B tile: im2col gather, n-major ----
#pragma unroll
        for (int i = tid; i < NTILE * 32; i += NT) {
            int n = i >> 5, kk = i & 31;
            int gk = k0 + kk, gn = nBase + n;
            __bf16 v = (__bf16)0.f;
            if (gk < K && gn < N) {
                int ci = gk / 25, r = gk % 25;
                int ky = r / 5, kx = r % 5;
                int b  = gn / (HOUT * WOUT), rr = gn % (HOUT * WOUT);
                int oh = rr / WOUT, ow = rr % WOUT;
                int ih = oh * 3 - 2 + ky, iw = ow * 3 - 2 + kx;
                if (ih >= 0 && ih < HIN && iw >= 0 && iw < WIN)
                    v = in[(((size_t)b * CIN + ci) * HIN + ih) * WIN + iw];
            }
            lB[i] = v;
        }
        wait_async_then_barrier();

        // ---- fragments: vectorized LDS reads, contiguous k order (k = 16*hl + s) in A and B ----
        // A lane frag: row m=l15, k in [hl*16,+16)  -> 2x b128
        const __bf16* ar = &lA[(wm * 16 + l15) * 32];
        v8bf a0 = *(const v8bf*)(ar + hl * 16);
        v8bf a1 = *(const v8bf*)(ar + hl * 16 + 8);
        v16bf afrag = __builtin_shufflevector(a0, a1,
            0, 1, 2, 3, 4, 5, 6, 7, 8, 9, 10, 11, 12, 13, 14, 15);
        // B lane frag: col n=l15, k in [hl*16,+16) -> contiguous 32B
        v16bf bfrag = *(const v16bf*)(&lB[(wn * 16 + l15) * 32 + hl * 16]);

        acc = __builtin_amdgcn_wmma_f32_16x16x32_bf16(
                  false, afrag, false, bfrag, (short)0, acc, false, false);
        __syncthreads();
    }

    // ---- epilogue: C/D layout n = lane%16, m = r + 8*(lane/16) ----
    int gn = nBase + wn * 16 + l15;
    if (gn < N) {
        int b  = gn / (HOUT * WOUT), rr = gn % (HOUT * WOUT);
        int oh = rr / WOUT, ow = rr % WOUT;
#pragma unroll
        for (int r = 0; r < 8; ++r) {
            int gm = mBase + wm * 16 + r + hl * 8;
            float v = acc[r] + bias[gm];
            if (HAS_BN) v = v * (bn_g[gm] * BN_RSQRT) + bn_b[gm];
            v = fmaxf(v, 0.f);
            size_t oidx = (((size_t)b * COUT + gm) * HOUT + oh) * WOUT + ow;
            if (OUT_BF16) ((__bf16*)outp)[oidx] = (__bf16)v;
            else          ((float*)outp)[oidx]  = v;
        }
    }
}

// ---------------------------------------------------------------------------
// Peripheral branch: adaptive max pool (2x3) + L2 norm + positional encoding
// ---------------------------------------------------------------------------
__device__ __forceinline__ float block_reduce_sum(float v, float* red, int tid)
{
    red[tid] = v;
    __syncthreads();
    for (int s = 128; s > 0; s >>= 1) {
        if (tid < s) red[tid] += red[tid + s];
        __syncthreads();
    }
    float r = red[0];
    __syncthreads();
    return r;
}

__global__ __launch_bounds__(256)
void periph_kernel(const float* __restrict__ fm, float* __restrict__ out, int B)
{
    __shared__ float red[256];
    const int cell = blockIdx.x / B, b = blockIdx.x % B;
    const int ri = cell / 3, ci = cell % 3;
    const int rs[2] = {0, 4},      re[2] = {4, 8};
    const int cs[3] = {0, 7, 15},  ce[3] = {8, 16, 23};
    const int tid = threadIdx.x;

    float pooled[3];
    float ssq = 0.f;
#pragma unroll
    for (int j = 0; j < 3; ++j) {
        int c = tid + j * 256;
        float m = -3.0e38f;
        for (int h = rs[ri]; h < re[ri]; ++h)
            for (int wv = cs[ci]; wv < ce[ci]; ++wv)
                m = fmaxf(m, fm[(((size_t)b * 768 + c) * 8 + h) * 23 + wv]);
        pooled[j] = m;
        ssq += m * m;
    }
    float nrm2 = block_reduce_sum(ssq, red, tid);
    float inv = 1.f / fmaxf(sqrtf(nrm2), 1e-12f);

    float avg_row = (ri + 0.5f) * 4.0f;
    float avg_col = (ci + 0.5f) * (23.0f / 3.0f);
    float nx = (avg_row / 22.f) * 0.01f;
    float ny = (avg_col / 7.f)  * 0.01f;
#pragma unroll
    for (int j = 0; j < 3; ++j) {
        int c = tid + j * 256;
        float pe = ((c & 1) == 0) ? nx : ny;
        out[((size_t)cell * B + b) * 768 + c] = pooled[j] * inv + pe;
    }
}

// ---------------------------------------------------------------------------
// Fovea MLP: fc1+LN+ReLU -> fc2+LN+ReLU -> fc3+sigmoid -> softmax(x10) + argmax
// ---------------------------------------------------------------------------
__global__ __launch_bounds__(256)
void fovea_mlp(const float* __restrict__ mem, const float* __restrict__ cls,
               const float* __restrict__ fc1w, const float* __restrict__ fc1b,
               const float* __restrict__ ln1g, const float* __restrict__ ln1b,
               const float* __restrict__ fc2w, const float* __restrict__ fc2b,
               const float* __restrict__ ln2g, const float* __restrict__ ln2b,
               const float* __restrict__ fc3w, const float* __restrict__ fc3b,
               float* __restrict__ sm, int* __restrict__ amax, int B)
{
    const int b = blockIdx.x, tid = threadIdx.x;
    __shared__ float h0[1536];
    __shared__ float h1[256];
    __shared__ float h2[128];
    __shared__ float red[256];
    __shared__ int   ridx[256];

    for (int i = tid; i < 768; i += 256) {
        h0[i]       = mem[(size_t)b * 768 + i];
        h0[768 + i] = cls[(size_t)b * 768 + i];
    }
    __syncthreads();

    float a = fc1b[tid];
    {
        const float* wr = fc1w + (size_t)tid * 1536;
        for (int k = 0; k < 1536; ++k) a = fmaf(wr[k], h0[k], a);
    }
    float mean = block_reduce_sum(a, red, tid) * (1.f / 256.f);
    float d = a - mean;
    float var = block_reduce_sum(d * d, red, tid) * (1.f / 256.f);
    a = d * rsqrtf(var + 1e-5f) * ln1g[tid] + ln1b[tid];
    h1[tid] = fmaxf(a, 0.f);
    __syncthreads();

    float a2 = 0.f;
    if (tid < 128) {
        a2 = fc2b[tid];
        const float* wr = fc2w + (size_t)tid * 256;
        for (int k = 0; k < 256; ++k) a2 = fmaf(wr[k], h1[k], a2);
    }
    float mean2 = block_reduce_sum(tid < 128 ? a2 : 0.f, red, tid) * (1.f / 128.f);
    float d2 = a2 - mean2;
    float var2 = block_reduce_sum(tid < 128 ? d2 * d2 : 0.f, red, tid) * (1.f / 128.f);
    if (tid < 128)
        h2[tid] = fmaxf(d2 * rsqrtf(var2 + 1e-5f) * ln2g[tid] + ln2b[tid], 0.f);
    __syncthreads();

    float z = -3.0e38f;
    if (tid < 184) {
        float a3 = fc3b[tid];
        const float* wr = fc3w + (size_t)tid * 128;
        for (int k = 0; k < 128; ++k) a3 = fmaf(wr[k], h2[k], a3);
        float sig = 1.f / (1.f + __expf(-a3));
        z = sig * 10.f;
    }
    red[tid] = z; ridx[tid] = tid;
    __syncthreads();
    for (int s = 128; s > 0; s >>= 1) {
        if (tid < s) {
            if (red[tid + s] > red[tid]) { red[tid] = red[tid + s]; ridx[tid] = ridx[tid + s]; }
        }
        __syncthreads();
    }
    float zmax = red[0];
    int   zarg = ridx[0];
    __syncthreads();
    float e = (tid < 184) ? __expf(z - zmax) : 0.f;
    float es = block_reduce_sum(e, red, tid);
    if (tid < 184) sm[(size_t)b * 184 + tid] = e / es;
    if (tid == 0)  amax[b] = zarg;
}

// ---------------------------------------------------------------------------
// Fovea feature: wf[b,c] = sum_hw fm[b,c,hw]*sm[b,hw]; L2 norm; argmax PE
// ---------------------------------------------------------------------------
__global__ __launch_bounds__(256)
void fovea_feat(const float* __restrict__ fm, const float* __restrict__ sm,
                const int* __restrict__ amax, float* __restrict__ out, int B)
{
    const int b = blockIdx.x, tid = threadIdx.x;
    __shared__ float wmap[184];
    __shared__ float red[256];
    for (int i = tid; i < 184; i += 256) wmap[i] = sm[(size_t)b * 184 + i];
    __syncthreads();

    float acc3[3];
    float ssq = 0.f;
#pragma unroll
    for (int j = 0; j < 3; ++j) {
        int c = tid + j * 256;
        const float* p = fm + ((size_t)b * 768 + c) * 184;
        float a = 0.f;
        for (int k = 0; k < 184; ++k) a = fmaf(p[k], wmap[k], a);
        acc3[j] = a;
        ssq += a * a;
    }
    float nrm2 = block_reduce_sum(ssq, red, tid);
    float inv = 1.f / fmaxf(sqrtf(nrm2), 1e-12f);

    int mp = amax[b];
    float mr = (float)(mp / 23), mc = (float)(mp % 23);
    float nx = (mr / 22.f) * 0.01f, ny = (mc / 7.f) * 0.01f;
#pragma unroll
    for (int j = 0; j < 3; ++j) {
        int c = tid + j * 256;
        float pe = ((c & 1) == 0) ? nx : ny;
        out[((size_t)6 * B + b) * 768 + c] = acc3[j] * inv + pe;
    }
}

// ---------------------------------------------------------------------------
// Launcher
// ---------------------------------------------------------------------------
extern "C" void kernel_launch(void* const* d_in, const int* in_sizes, int n_in,
                              void* d_out, int out_size, void* d_ws, size_t ws_size,
                              hipStream_t stream)
{
    const float* imgs = (const float*)d_in[0];
    const float* mem  = (const float*)d_in[1];
    const float* cls  = (const float*)d_in[2];
    const float* c1w = (const float*)d_in[3];  const float* c1b = (const float*)d_in[4];
    const float* c2w = (const float*)d_in[5];  const float* c2b = (const float*)d_in[6];
    const float* c3w = (const float*)d_in[7];  const float* c3b = (const float*)d_in[8];
    const float* c4w = (const float*)d_in[9];  const float* c4b = (const float*)d_in[10];
    const float* g1  = (const float*)d_in[11]; const float* be1 = (const float*)d_in[12];
    const float* g2  = (const float*)d_in[13]; const float* be2 = (const float*)d_in[14];
    const float* g3  = (const float*)d_in[15]; const float* be3 = (const float*)d_in[16];
    const float* fc1w = (const float*)d_in[17]; const float* fc1b = (const float*)d_in[18];
    const float* ln1g = (const float*)d_in[19]; const float* ln1b = (const float*)d_in[20];
    const float* fc2w = (const float*)d_in[21]; const float* fc2b = (const float*)d_in[22];
    const float* ln2g = (const float*)d_in[23]; const float* ln2b = (const float*)d_in[24];
    const float* fc3w = (const float*)d_in[25]; const float* fc3b = (const float*)d_in[26];
    float* out = (float*)d_out;

    const int B = in_sizes[1] / 768;   // 16

    size_t off = 0;
    auto alloc = [&](size_t bytes) -> void* {
        void* p = (char*)d_ws + off;
        off += (bytes + 255) & ~(size_t)255;
        return p;
    };
    __bf16* y1  = (__bf16*)alloc((size_t)B * 16  * 200 * 600 * sizeof(__bf16));
    __bf16* y2  = (__bf16*)alloc((size_t)B * 64  * 67  * 200 * sizeof(__bf16));
    __bf16* y3  = (__bf16*)alloc((size_t)B * 256 * 23  * 67  * sizeof(__bf16));
    float*  fm  = (float*) alloc((size_t)B * 768 * 8   * 23  * sizeof(float));
    float*  sm  = (float*) alloc((size_t)B * 184 * sizeof(float));
    int*  amax  = (int*)   alloc((size_t)B * sizeof(int));
    __bf16* wb2 = (__bf16*)alloc((size_t)64  * 416  * sizeof(__bf16));  // K=400 ->Kp=416
    __bf16* wb3 = (__bf16*)alloc((size_t)256 * 1600 * sizeof(__bf16));  // K=1600
    __bf16* wb4 = (__bf16*)alloc((size_t)768 * 6400 * sizeof(__bf16));  // K=6400

    // 0) pre-convert conv weights to padded bf16
    convert_weights<<<(64 * 416 + 255) / 256, 256, 0, stream>>>(c2w, wb2, 64, 400, 416);
    convert_weights<<<(256 * 1600 + 255) / 256, 256, 0, stream>>>(c3w, wb3, 256, 1600, 1600);
    convert_weights<<<(768 * 6400 + 255) / 256, 256, 0, stream>>>(c4w, wb4, 768, 6400, 6400);

    // 1) conv1 (fused bilinear resize + tile + BN + ReLU)
    {
        long npos = (long)B * 200 * 600;
        dim3 grid((unsigned)((npos + 255) / 256));
        conv1_fused<<<grid, 256, 0, stream>>>(imgs, c1w, c1b, g1, be1, y1, B);
    }
    // 2) conv2: 16->64, 200x600 -> 67x200 (WMMA bf16)
    {
        int N = B * 67 * 200;
        dim3 grid((unsigned)((N + 31) / 32), 1);
        conv_wmma<64, 32, true, true, 16, 200, 600, 64, 67, 200>
            <<<grid, 256, 0, stream>>>(y1, wb2, c2b, g2, be2, (void*)y2, B);
    }
    // 3) conv3: 64->256, 67x200 -> 23x67
    {
        int N = B * 23 * 67;
        dim3 grid((unsigned)((N + 31) / 32), 4);
        conv_wmma<64, 32, true, true, 64, 67, 200, 256, 23, 67>
            <<<grid, 256, 0, stream>>>(y2, wb3, c3b, g3, be3, (void*)y3, B);
    }
    // 4) conv4: 256->768, 23x67 -> 8x23 (no BN, f32 out)
    {
        int N = B * 8 * 23;
        dim3 grid((unsigned)((N + 31) / 32), 12);
        conv_wmma<64, 32, false, false, 256, 23, 67, 768, 8, 23>
            <<<grid, 256, 0, stream>>>(y3, wb4, c4b, nullptr, nullptr, (void*)fm, B);
    }
    // 5) peripheral cells 0..5
    periph_kernel<<<6 * B, 256, 0, stream>>>(fm, out, B);
    // 6) fovea MLP -> softmax weights + argmax
    fovea_mlp<<<B, 256, 0, stream>>>(mem, cls, fc1w, fc1b, ln1g, ln1b,
                                     fc2w, fc2b, ln2g, ln2b, fc3w, fc3b, sm, amax, B);
    // 7) fovea feature -> cell 6
    fovea_feat<<<B, 256, 0, stream>>>(fm, sm, amax, out, B);
}
